// CortexBlock_63986422775910
// MI455X (gfx1250) — matrix operand order repaired and checked by reading, not compiled, not run on hardware
//
#include <hip/hip_runtime.h>

// ---------------- problem constants ----------------
constexpr int B_   = 4;
constexpr int T_   = 2048;
constexpr int D_   = 1024;   // d_model
constexpr int H_   = 16;     // heads
constexpr int DH_  = 64;     // d_head
constexpr int C_   = 1024;   // H*DH
constexpr int M_   = B_ * T_;    // 8192 rows
constexpr int KC_  = 4;      // conv kernel

typedef unsigned short u16;
typedef __bf16 bf16x8  __attribute__((ext_vector_type(8)));
typedef __bf16 bf16x16 __attribute__((ext_vector_type(16)));
typedef float  f32x8   __attribute__((ext_vector_type(8)));
typedef u16    u16x8   __attribute__((ext_vector_type(8)));

#if defined(__HIP_DEVICE_COMPILE__) && __has_builtin(__builtin_amdgcn_global_load_async_to_lds_b128)
#define USE_ASYNC_LDS 1
typedef int v4i __attribute__((ext_vector_type(4)));
typedef __attribute__((address_space(1))) v4i as1_v4i;   // global
typedef __attribute__((address_space(3))) v4i as3_v4i;   // LDS
#else
#define USE_ASYNC_LDS 0
#endif

static __device__ __forceinline__ u16 f2bf(float f) {
    unsigned u = __float_as_uint(f);
    unsigned r = (u + 0x7FFFu + ((u >> 16) & 1u)) >> 16;   // RNE
    return (u16)r;
}
static __device__ __forceinline__ float silu(float x) {
    return x / (1.f + __expf(-x));
}

// ---------------- fp32 -> bf16 convert ----------------
__global__ void f32_to_bf16_kernel(const float* __restrict__ in,
                                   u16* __restrict__ out, size_t n) {
    size_t i = (size_t)blockIdx.x * blockDim.x + threadIdx.x;
    size_t stride = (size_t)gridDim.x * blockDim.x;
    for (; i < n; i += stride) out[i] = f2bf(in[i]);
}

// ---------------- bf16 WMMA GEMM: C[M,N] = A[M,K] * B[N,K]^T ----------------
constexpr int BM = 128, BN = 128, BK = 32;
constexpr int LDT = BK + 8;   // padded LDS row (u16 units) -> 80B rows

__global__ __launch_bounds__(256)
void gemm_bf16_wmma(const u16* __restrict__ A, const u16* __restrict__ Bw,
                    float* __restrict__ Cout, int M, int N, int Kd) {
    __shared__ u16 As[2][BM * LDT];
    __shared__ u16 Bs[2][BN * LDT];

    const int tid  = threadIdx.x;
    const int lane = tid & 31;
    const int wave = tid >> 5;
    const int wm   = wave & 3;        // row group: 32 rows each
    const int wn   = wave >> 2;       // col group: 64 cols each
    const int l    = lane & 15;
    const int hi   = lane >> 4;
    const int bm   = blockIdx.x * BM;
    const int bn   = blockIdx.y * BN;
    const int NT   = Kd / BK;

    auto load_tile = [&](int kt, int buf) {
        const int k0 = kt * BK;
#pragma unroll
        for (int i = 0; i < 2; ++i) {
            int chunk = tid * 2 + i;          // 512 chunks of 8 halfs
            int row   = chunk >> 2;           // BK/8 = 4 chunks per row
            int kc    = (chunk & 3) * 8;
            const u16* gA = A  + (size_t)(bm + row) * Kd + k0 + kc;
            const u16* gB = Bw + (size_t)(bn + row) * Kd + k0 + kc;
            u16* lA = &As[buf][row * LDT + kc];
            u16* lB = &Bs[buf][row * LDT + kc];
#if USE_ASYNC_LDS
            // 16B/lane DMA straight into LDS (ASYNCcnt-tracked, no VGPR staging)
            __builtin_amdgcn_global_load_async_to_lds_b128(
                (as1_v4i*)gA, (as3_v4i*)lA, 0, 0);
            __builtin_amdgcn_global_load_async_to_lds_b128(
                (as1_v4i*)gB, (as3_v4i*)lB, 0, 0);
#else
            *(u16x8*)lA = *(const u16x8*)gA;
            *(u16x8*)lB = *(const u16x8*)gB;
#endif
        }
    };

    f32x8 acc[2][4] = {};
    load_tile(0, 0);

    for (int kt = 0; kt < NT; ++kt) {
        const int buf = kt & 1;
#if USE_ASYNC_LDS
        asm volatile("s_wait_asynccnt 0x0" ::: "memory");
#endif
        __syncthreads();
        if (kt + 1 < NT) load_tile(kt + 1, buf ^ 1);

        // A fragments: 16x32 bf16, lane<16: row l, K {0..7,16..23}; lane>=16: K {8..15,24..31}
        bf16x16 af[2];
#pragma unroll
        for (int r = 0; r < 2; ++r) {
            int row = wm * 32 + r * 16 + l;
            const u16* p = &As[buf][row * LDT + hi * 8];
            bf16x8 lo  = *(const bf16x8*)p;
            bf16x8 hi8 = *(const bf16x8*)(p + 16);
#pragma unroll
            for (int x = 0; x < 8; ++x) { af[r][x] = lo[x]; af[r][x + 8] = hi8[x]; }
        }
        // B fragments: 32x16 bf16, lane<16: col l holds K {0..7,16..23}; lane>=16: K {8..15,24..31}
        bf16x16 bf[4];
#pragma unroll
        for (int c = 0; c < 4; ++c) {
            int col = wn * 64 + c * 16 + l;
            const u16* p = &Bs[buf][col * LDT + hi * 8];
            bf16x8 lo  = *(const bf16x8*)p;
            bf16x8 hi8 = *(const bf16x8*)(p + 16);
#pragma unroll
            for (int x = 0; x < 8; ++x) { bf[c][x] = lo[x]; bf[c][x + 8] = hi8[x]; }
        }
#pragma unroll
        for (int r = 0; r < 2; ++r)
#pragma unroll
            for (int c = 0; c < 4; ++c)
                acc[r][c] = __builtin_amdgcn_wmma_f32_16x16x32_bf16(
                    false, af[r], false, bf[c], (short)0, acc[r][c], false, false);
    }

    // epilogue: D layout -> lane<16: Vj = D[j, l]; lane>=16: Vj = D[j+8, l]
#pragma unroll
    for (int r = 0; r < 2; ++r)
#pragma unroll
        for (int c = 0; c < 4; ++c) {
            int row0 = bm + wm * 32 + r * 16 + hi * 8;
            int col  = bn + wn * 64 + c * 16 + l;
#pragma unroll
            for (int j = 0; j < 8; ++j)
                Cout[(size_t)(row0 + j) * N + col] = acc[r][c][j];
        }
}

// ---------------- alpha/beta projections ----------------
__global__ __launch_bounds__(256)
void alphabeta_kernel(const float* __restrict__ X,
                      const float* __restrict__ Wa, const float* __restrict__ ba,
                      const float* __restrict__ Wb, const float* __restrict__ bb,
                      float* __restrict__ alpha, float* __restrict__ beta) {
    int m = blockIdx.x;
    int wave = threadIdx.x >> 5, lane = threadIdx.x & 31;
    const float* x = X + (size_t)m * D_;
#pragma unroll
    for (int o = 0; o < 4; ++o) {
        int idx = wave * 4 + o;                       // 0..31 (16 alpha + 16 beta)
        const float* W = (idx < 16) ? (Wa + (size_t)idx * D_)
                                    : (Wb + (size_t)(idx - 16) * D_);
        float s = 0.f;
        for (int kk = lane; kk < D_; kk += 32) s += x[kk] * W[kk];
#pragma unroll
        for (int off = 16; off; off >>= 1) s += __shfl_xor(s, off);
        if (lane == 0) {
            if (idx < 16) alpha[(size_t)m * H_ + idx] = 1.f / (1.f + __expf(-(s + ba[idx])));
            else          beta [(size_t)m * H_ + idx - 16] = 1.f / (1.f + __expf(-(s + bb[idx - 16])));
        }
    }
}

// ---------------- causal depthwise conv + SiLU + k L2-norm ----------------
__global__ __launch_bounds__(256)
void conv_silu_norm_kernel(const float* __restrict__ Qr, const float* __restrict__ Kr,
                           const float* __restrict__ Vr,
                           const float* __restrict__ wq, const float* __restrict__ bq,
                           const float* __restrict__ wk, const float* __restrict__ bk,
                           const float* __restrict__ wv, const float* __restrict__ bv,
                           float* __restrict__ Qc, float* __restrict__ Kc,
                           float* __restrict__ Vc) {
    int m = blockIdx.x;
    int b = m / T_, t = m % T_;
    int c0 = threadIdx.x * 4;                  // 4 contiguous channels / thread
    float q[4] = {}, k[4] = {}, v[4] = {};
#pragma unroll
    for (int j = 0; j < KC_; ++j) {
        int tt = t - (KC_ - 1) + j;
        if (tt < 0) continue;
        size_t off = ((size_t)(b * T_ + tt)) * C_ + c0;
        float4 xq = *(const float4*)(Qr + off);
        float4 xk = *(const float4*)(Kr + off);
        float4 xv = *(const float4*)(Vr + off);
        const float* pq = &xq.x; const float* pk = &xk.x; const float* pv = &xv.x;
#pragma unroll
        for (int i = 0; i < 4; ++i) {
            q[i] += wq[(c0 + i) * KC_ + j] * pq[i];
            k[i] += wk[(c0 + i) * KC_ + j] * pk[i];
            v[i] += wv[(c0 + i) * KC_ + j] * pv[i];
        }
    }
    float ss = 0.f;
#pragma unroll
    for (int i = 0; i < 4; ++i) {
        q[i] = silu(q[i] + bq[c0 + i]);
        k[i] = silu(k[i] + bk[c0 + i]);
        v[i] = silu(v[i] + bv[c0 + i]);
        ss += k[i] * k[i];
    }
    // 64-channel head = 16 consecutive lanes; reduce within group of 16
#pragma unroll
    for (int off = 8; off; off >>= 1) ss += __shfl_xor(ss, off, 16);
    float inv = 1.f / fmaxf(sqrtf(ss), 1e-12f);
    size_t o = ((size_t)m) * C_ + c0;
    float4 oq, ok, ov;
    oq.x = q[0]; oq.y = q[1]; oq.z = q[2]; oq.w = q[3];
    ok.x = k[0] * inv; ok.y = k[1] * inv; ok.z = k[2] * inv; ok.w = k[3] * inv;
    ov.x = v[0]; ov.y = v[1]; ov.z = v[2]; ov.w = v[3];
    *(float4*)(Qc + o) = oq;
    *(float4*)(Kc + o) = ok;
    *(float4*)(Vc + o) = ov;
}

// ---------------- sequential delta-rule scan (64x64 state per (b,h)) ----------------
__global__ __launch_bounds__(64)
void scan_kernel(const float* __restrict__ Qc, const float* __restrict__ Kc,
                 const float* __restrict__ Vc,
                 const float* __restrict__ alpha, const float* __restrict__ beta,
                 float* __restrict__ Y) {
    int bh = blockIdx.x;
    int b = bh / H_, h = bh % H_;
    int i = threadIdx.x;                 // state row
    __shared__ float sk[2][DH_], sq[2][DH_];
    float S[DH_];
#pragma unroll
    for (int j = 0; j < DH_; ++j) S[j] = 0.f;

    for (int t = 0; t < T_; ++t) {
        size_t base = ((size_t)(b * T_ + t)) * C_ + h * DH_;
        float kv = Kc[base + i], qv = Qc[base + i], vv = Vc[base + i];
        int bufb = t & 1;
        sk[bufb][i] = kv;
        sq[bufb][i] = qv;
        float a  = alpha[(size_t)(b * T_ + t) * H_ + h];
        float bt = beta [(size_t)(b * T_ + t) * H_ + h];
        __syncthreads();
        float vold = 0.f;
#pragma unroll
        for (int j = 0; j < DH_; ++j) vold += S[j] * sk[bufb][j];
        float coef = bt * (vv - a * vold);
        float o = 0.f;
#pragma unroll
        for (int j = 0; j < DH_; ++j) {
            float s = fminf(5.f, fmaxf(-5.f, a * S[j] + coef * sk[bufb][j]));
            S[j] = s;
            o += s * sq[bufb][j];
        }
        Y[base + i] = o;
    }
}

// ---------------- gate: y = scan * silu(G + bg), emit bf16 ----------------
__global__ void gate_kernel(const float* __restrict__ Y, const float* __restrict__ G,
                            const float* __restrict__ bg, u16* __restrict__ Ybf,
                            size_t n) {
    size_t i = (size_t)blockIdx.x * blockDim.x + threadIdx.x;
    size_t stride = (size_t)gridDim.x * blockDim.x;
    for (; i < n; i += stride) {
        int c = (int)(i % C_);
        float g = silu(G[i] + bg[c]);
        Ybf[i] = f2bf(Y[i] * g);
    }
}

// ---------------- launcher ----------------
extern "C" void kernel_launch(void* const* d_in, const int* in_sizes, int n_in,
                              void* d_out, int out_size, void* d_ws, size_t ws_size,
                              hipStream_t stream) {
    const float* hidden = (const float*)d_in[0];
    const float* Wq  = (const float*)d_in[1];
    const float* Wk  = (const float*)d_in[2];
    const float* Wv  = (const float*)d_in[3];
    const float* Wa  = (const float*)d_in[4];
    const float* ba  = (const float*)d_in[5];
    const float* Wb  = (const float*)d_in[6];
    const float* bb  = (const float*)d_in[7];
    const float* cqw = (const float*)d_in[8];
    const float* cqb = (const float*)d_in[9];
    const float* ckw = (const float*)d_in[10];
    const float* ckb = (const float*)d_in[11];
    const float* cvw = (const float*)d_in[12];
    const float* cvb = (const float*)d_in[13];
    const float* Wg  = (const float*)d_in[14];
    const float* bg  = (const float*)d_in[15];
    const float* Wo  = (const float*)d_in[16];
    float* out = (float*)d_out;

    char* ws = (char*)d_ws;
    size_t off = 0;
    auto take = [&](size_t bytes) -> char* {
        char* p = ws + off;
        off += (bytes + 255) & ~(size_t)255;
        return p;
    };
    u16*   Xbf   = (u16*)  take((size_t)M_ * D_ * 2);
    u16*   Wqb   = (u16*)  take((size_t)C_ * D_ * 2);
    u16*   Wkb   = (u16*)  take((size_t)C_ * D_ * 2);
    u16*   Wvb   = (u16*)  take((size_t)C_ * D_ * 2);
    u16*   Wgb   = (u16*)  take((size_t)C_ * D_ * 2);
    u16*   Wob   = (u16*)  take((size_t)D_ * C_ * 2);
    float* Qraw  = (float*)take((size_t)M_ * C_ * 4);
    float* Kraw  = (float*)take((size_t)M_ * C_ * 4);
    float* Vraw  = (float*)take((size_t)M_ * C_ * 4);
    float* Graw  = (float*)take((size_t)M_ * C_ * 4);
    float* Qc    = (float*)take((size_t)M_ * C_ * 4);
    float* Kc    = (float*)take((size_t)M_ * C_ * 4);
    float* Vc    = (float*)take((size_t)M_ * C_ * 4);
    float* alphaP= (float*)take((size_t)M_ * H_ * 4);
    float* betaP = (float*)take((size_t)M_ * H_ * 4);
    float* Yscan = Qraw;                 // dead after conv
    u16*   Ybf   = (u16*)Kraw;           // dead after conv

    // 1) bf16 conversions
    f32_to_bf16_kernel<<<4096, 256, 0, stream>>>(hidden, Xbf, (size_t)M_ * D_);
    f32_to_bf16_kernel<<<1024, 256, 0, stream>>>(Wq, Wqb, (size_t)C_ * D_);
    f32_to_bf16_kernel<<<1024, 256, 0, stream>>>(Wk, Wkb, (size_t)C_ * D_);
    f32_to_bf16_kernel<<<1024, 256, 0, stream>>>(Wv, Wvb, (size_t)C_ * D_);
    f32_to_bf16_kernel<<<1024, 256, 0, stream>>>(Wg, Wgb, (size_t)C_ * D_);
    f32_to_bf16_kernel<<<1024, 256, 0, stream>>>(Wo, Wob, (size_t)D_ * C_);

    // 2) projection GEMMs (WMMA bf16 -> fp32)
    dim3 ggrid(M_ / BM, C_ / BN);
    gemm_bf16_wmma<<<ggrid, 256, 0, stream>>>(Xbf, Wqb, Qraw, M_, C_, D_);
    gemm_bf16_wmma<<<ggrid, 256, 0, stream>>>(Xbf, Wkb, Kraw, M_, C_, D_);
    gemm_bf16_wmma<<<ggrid, 256, 0, stream>>>(Xbf, Wvb, Vraw, M_, C_, D_);
    gemm_bf16_wmma<<<ggrid, 256, 0, stream>>>(Xbf, Wgb, Graw, M_, C_, D_);

    // 3) alpha/beta (fp32)
    alphabeta_kernel<<<M_, 256, 0, stream>>>(hidden, Wa, ba, Wb, bb, alphaP, betaP);

    // 4) conv + silu + k-norm
    conv_silu_norm_kernel<<<M_, 256, 0, stream>>>(Qraw, Kraw, Vraw,
                                                  cqw, cqb, ckw, ckb, cvw, cvb,
                                                  Qc, Kc, Vc);

    // 5) sequential scan
    scan_kernel<<<B_ * H_, 64, 0, stream>>>(Qc, Kc, Vc, alphaP, betaP, Yscan);

    // 6) gate multiply, emit bf16
    gate_kernel<<<8192, 256, 0, stream>>>(Yscan, Graw, bg, Ybf, (size_t)M_ * C_);

    // 7) output GEMM: out[m,d] = sum_c y[m,c] * Wo[d,c]
    dim3 ogrid(M_ / BM, D_ / BN);
    gemm_bf16_wmma<<<ogrid, 256, 0, stream>>>(Ybf, Wob, out, M_, D_, C_);

    (void)in_sizes; (void)n_in; (void)out_size; (void)ws_size;
}
